// RNNsym_23021024706899
// MI455X (gfx1250) — matrix operand order repaired and checked by reading, not compiled, not run on hardware
//
#include <hip/hip_runtime.h>
#include <hip/hip_bf16.h>
#include <math.h>

// ---------------------------------------------------------------------------
// Symmetrized 2-layer RNN wavefunction (RNNsym) for MI455X / gfx1250.
//   orbit permutations == cyclic shifts -> pure index arithmetic.
//   Per wave: 16 sequences resident; per step 3 GEMMs [16x64]x[64x64] done as
//   v_wmma_f32_16x16x32_f16 (f32 accum), elu in f32, state relayout via LDS.
// ---------------------------------------------------------------------------

#define LSEQ 128
#define HID 64
#define KORB 128
#define NB 64
#define WAVES 4          // waves per workgroup
#define ROWS_PER_WG 64   // 4 waves * 16 rows
#define STG_STRIDE 72    // padded halves per staged row (bank-conflict friendly)
#define STG_SIZE (16 * STG_STRIDE)
#define BFRAG_LANE 24    // padded halves per lane in frag-major weight storage
#define BFRAG_SIZE (32 * BFRAG_LANE)   // 768 halves per (mat,nt,kc) fragment

typedef __attribute__((ext_vector_type(16))) _Float16 v16h;
typedef __attribute__((ext_vector_type(8)))  _Float16 v8h;
typedef __attribute__((ext_vector_type(8)))  float    v8f;

__device__ __forceinline__ v8f wmma_f16(v16h a, v16h b, v8f c) {
    // D = A(16x32 f16) x B(32x16 f16) + C(16x16 f32)
    return __builtin_amdgcn_wmma_f32_16x16x32_f16(
        /*neg_a=*/false, a, /*neg_b=*/false, b,
        /*c_mod=*/(short)0, c, /*reuse_a=*/false, /*reuse_b=*/false);
}

// A fragment (16x32, f16) from row-major staging: lane holds row ln; K halves:
//   elems 0..7  -> cols kc*32 + 8*hi + [0..8)
//   elems 8..15 -> cols kc*32 + 16 + 8*hi + [0..8)
__device__ __forceinline__ v16h load_afrag(const _Float16* st, int ln, int hi, int kc) {
    const int base = ln * STG_STRIDE + kc * 32 + 8 * hi;
    v8h p0 = *(const v8h*)(st + base);
    v8h p1 = *(const v8h*)(st + base + 16);
    return __builtin_shufflevector(p0, p1, 0, 1, 2, 3, 4, 5, 6, 7,
                                   8, 9, 10, 11, 12, 13, 14, 15);
}

// B fragment (32x16, f16) from pre-swizzled frag-major LDS region.
__device__ __forceinline__ v16h load_bfrag(const _Float16* region, int lane) {
    const _Float16* p = region + lane * BFRAG_LANE;
    v8h p0 = *(const v8h*)(p);
    v8h p1 = *(const v8h*)(p + 8);
    return __builtin_shufflevector(p0, p1, 0, 1, 2, 3, 4, 5, 6, 7,
                                   8, 9, 10, 11, 12, 13, 14, 15);
}

__global__ __launch_bounds__(128)
void rnn_sym_kernel(const int* __restrict__ x,
                    const float* __restrict__ Wi0, const float* __restrict__ Wc0,
                    const float* __restrict__ bc0, const float* __restrict__ Wi1,
                    const float* __restrict__ Wc1, const float* __restrict__ bc1,
                    const float* __restrict__ Wout, const float* __restrict__ bout,
                    double* __restrict__ partials) {
    // wfrag: 3 mats (Wc0, Wi1, Wc1) x 4 ntiles x 2 kchunks, frag-major B layout
    __shared__ __align__(32) _Float16 wfrag[3 * 4 * 2 * BFRAG_SIZE];
    __shared__ __align__(32) _Float16 wofrag[2 * BFRAG_SIZE];        // padded Wout
    __shared__ __align__(32) _Float16 stage0[WAVES * STG_SIZE];      // h0 staging
    __shared__ __align__(32) _Float16 stage1[WAVES * STG_SIZE];      // h1 staging
    __shared__ int    xb[LSEQ];
    __shared__ float  logx[WAVES][16][2];
    __shared__ double part[WAVES];

    const int tid  = threadIdx.x;
    const int bidx = blockIdx.x;       // = b*2 + half
    const int b    = bidx >> 1;
    const int hwg  = bidx & 1;         // which half of the 128-orbit

    // ---- one-time cooperative init ---------------------------------------
    if (tid < LSEQ) xb[tid] = x[b * LSEQ + tid];

    // Pre-swizzle recurrent weights into B-fragment layout (f32 -> f16):
    //   lane l holds column n = nt*16 + (l&15); K = kc*32 + (e&7) + 16*(e>=8) + 8*(l>=16)
    for (int i = tid; i < 3 * 4 * 2 * 32 * 16; i += 128) {
        const int e    = i & 15;
        const int lane = (i >> 4) & 31;
        const int kc   = (i >> 9) & 1;
        const int nt   = (i >> 10) & 3;
        const int mat  = i >> 12;
        const int k    = kc * 32 + (e & 7) + ((e >> 3) << 4) + ((lane >> 4) << 3);
        const int n    = nt * 16 + (lane & 15);
        const float* W = (mat == 0) ? Wc0 : ((mat == 1) ? Wi1 : Wc1);
        wfrag[((mat * 4 + nt) * 2 + kc) * BFRAG_SIZE + lane * BFRAG_LANE + e] =
            (_Float16)W[k * HID + n];
    }
    // Wout padded to 64x16 (cols >=2 are zero)
    for (int i = tid; i < 2 * 32 * 16; i += 128) {
        const int e    = i & 15;
        const int lane = (i >> 4) & 31;
        const int kc   = i >> 9;
        const int k    = kc * 32 + (e & 7) + ((e >> 3) << 4) + ((lane >> 4) << 3);
        const int n    = lane & 15;
        wofrag[kc * BFRAG_SIZE + lane * BFRAG_LANE + e] =
            (n < 2) ? (_Float16)Wout[k * 2 + n] : (_Float16)0.f;
    }
    __syncthreads();

    // ---- per-thread constants --------------------------------------------
    const int w  = tid >> 5;           // wave id in WG
    const int l  = tid & 31;           // lane
    const int hi = l >> 4;
    const int ln = l & 15;
    const int rowbase = hwg * 64 + w * 16;   // orbit-shift k of local row 0

    float bc0v[4], bc1v[4], w0a[4], w0d[4];
#pragma unroll
    for (int nt = 0; nt < 4; ++nt) {
        const int col = nt * 16 + ln;
        bc0v[nt] = bc0[col];
        bc1v[nt] = bc1[col];
        w0a[nt]  = Wi0[col];                 // Wi0[0][col]
        w0d[nt]  = Wi0[HID + col] - w0a[nt]; // Wi0[1][col] - Wi0[0][col]
    }
    const float bout0 = bout[0], bout1 = bout[1];

    _Float16* st0 = stage0 + w * STG_SIZE;
    _Float16* st1 = stage1 + w * STG_SIZE;

    // hidden states as resident A fragments (zero initial state)
    v16h a0_0 = {}, a0_1 = {}, a1_0 = {}, a1_1 = {};
    float S = 0.f;   // sum of per-step log-probs for row `ln` (lanes 0..15)

    for (int t = 0; t < LSEQ; ++t) {
        const float sPrev = (t > 0) ? 1.f : 0.f;
        float xpf[8];
#pragma unroll
        for (int r = 0; r < 8; ++r) {
            const int m = r + 8 * hi;
            xpf[r] = (float)xb[(t - 1 + rowbase + m) & (LSEQ - 1)];
        }

        // ---- layer 0: h0n = elu(h0 @ Wc0 + bc0 + onehot(xprev) @ Wi0) ----
#pragma unroll
        for (int nt = 0; nt < 4; ++nt) {
            v8f acc = {};
            acc = wmma_f16(a0_0, load_bfrag(wfrag + ((0 * 4 + nt) * 2 + 0) * BFRAG_SIZE, l), acc);
            acc = wmma_f16(a0_1, load_bfrag(wfrag + ((0 * 4 + nt) * 2 + 1) * BFRAG_SIZE, l), acc);
#pragma unroll
            for (int r = 0; r < 8; ++r) {
                const int m = r + 8 * hi;
                float v = acc[r] + bc0v[nt] + sPrev * (w0a[nt] + w0d[nt] * xpf[r]);
                v = (v > 0.f) ? v : (__expf(v) - 1.f);
                st0[m * STG_STRIDE + nt * 16 + ln] = (_Float16)v;
            }
        }
        a0_0 = load_afrag(st0, ln, hi, 0);   // h0n in A layout (reused next step)
        a0_1 = load_afrag(st0, ln, hi, 1);

        // ---- layer 1: h1n = elu(h1 @ Wc1 + bc1 + h0n @ Wi1) --------------
#pragma unroll
        for (int nt = 0; nt < 4; ++nt) {
            v8f acc = {};
            acc = wmma_f16(a1_0, load_bfrag(wfrag + ((2 * 4 + nt) * 2 + 0) * BFRAG_SIZE, l), acc);
            acc = wmma_f16(a1_1, load_bfrag(wfrag + ((2 * 4 + nt) * 2 + 1) * BFRAG_SIZE, l), acc);
            acc = wmma_f16(a0_0, load_bfrag(wfrag + ((1 * 4 + nt) * 2 + 0) * BFRAG_SIZE, l), acc);
            acc = wmma_f16(a0_1, load_bfrag(wfrag + ((1 * 4 + nt) * 2 + 1) * BFRAG_SIZE, l), acc);
#pragma unroll
            for (int r = 0; r < 8; ++r) {
                const int m = r + 8 * hi;
                float v = acc[r] + bc1v[nt];
                v = (v > 0.f) ? v : (__expf(v) - 1.f);
                st1[m * STG_STRIDE + nt * 16 + ln] = (_Float16)v;
            }
        }
        a1_0 = load_afrag(st1, ln, hi, 0);   // h1n (reused next step for Wc1)
        a1_1 = load_afrag(st1, ln, hi, 1);

        // ---- logits = h1n @ Wout (padded 64x16, only N=0,1 valid) --------
        v8f al = {};
        al = wmma_f16(a1_0, load_bfrag(wofrag + 0 * BFRAG_SIZE, l), al);
        al = wmma_f16(a1_1, load_bfrag(wofrag + 1 * BFRAG_SIZE, l), al);
        if (ln < 2) {
#pragma unroll
            for (int r = 0; r < 8; ++r) logx[w][r + 8 * hi][ln] = al[r];
        }
        // same-wave LDS in-order: values visible; lanes 0..15 own row ln
        {
            const float lg0 = logx[w][ln][0] + bout0;
            const float lg1 = logx[w][ln][1] + bout1;
            const int   xt  = xb[(t + rowbase + ln) & (LSEQ - 1)];
            const float mx  = fmaxf(lg0, lg1);
            const float lse = mx + __logf(__expf(lg0 - mx) + __expf(lg1 - mx));
            S += (xt ? lg1 : lg0) - lse;     // garbage in lanes >=16, masked later
        }
    }

    // ---- orbit-mean partials (f64, deterministic: one slot per WG) --------
    double e = (ln < 16 && hi == 0) ? exp((double)S) : 0.0;
#pragma unroll
    for (int off = 16; off > 0; off >>= 1) e += __shfl_xor(e, off);
    if (l == 0) part[w] = e;
    __syncthreads();
    if (tid == 0) partials[bidx] = part[0] + part[1] + part[2] + part[3];
}

__global__ void rnn_sym_finalize(const double* __restrict__ partials,
                                 float* __restrict__ out) {
    const int b = threadIdx.x;
    if (b < NB) {
        const double s = partials[2 * b] + partials[2 * b + 1];
        out[b] = (float)(0.5 * log(s * (1.0 / (double)KORB)));
    }
}

extern "C" void kernel_launch(void* const* d_in, const int* in_sizes, int n_in,
                              void* d_out, int out_size, void* d_ws, size_t ws_size,
                              hipStream_t stream) {
    (void)in_sizes; (void)n_in; (void)out_size; (void)ws_size;
    const int*   x    = (const int*)d_in[0];
    // d_in[1] = orbit (cyclic permutations) -- replaced by index arithmetic
    const float* Wi0  = (const float*)d_in[2];
    const float* Wc0  = (const float*)d_in[3];
    const float* bc0  = (const float*)d_in[4];
    const float* Wi1  = (const float*)d_in[5];
    const float* Wc1  = (const float*)d_in[6];
    const float* bc1  = (const float*)d_in[7];
    const float* Wout = (const float*)d_in[8];
    const float* bout = (const float*)d_in[9];
    double* partials  = (double*)d_ws;        // 128 doubles
    float*  out       = (float*)d_out;        // 64 floats

    rnn_sym_kernel<<<dim3(NB * 2), dim3(128), 0, stream>>>(
        x, Wi0, Wc0, bc0, Wi1, Wc1, bc1, Wout, bout, partials);
    rnn_sym_finalize<<<dim3(1), dim3(64), 0, stream>>>(partials, out);
}
